// BertUnpadSelfAttentionWithExtras_37726992728254
// MI455X (gfx1250) — matrix operand order repaired and checked by reading, not compiled, not run on hardware
//
#include <hip/hip_runtime.h>
#include <hip/hip_bf16.h>

typedef __attribute__((ext_vector_type(16))) _Float16 v16h;
typedef __attribute__((ext_vector_type(8)))  _Float16 v8h;
typedef __attribute__((ext_vector_type(4)))  _Float16 v4h;
typedef __attribute__((ext_vector_type(8)))  float    v8f;

union FragAB { v16h v; v8h h[2]; };

#define DIMQ   768
#define QKVDIM 2304
#define NH     12
#define HD     64
#define LB     512
#define SB     1024
#define NB     4
#define NKSTEP 24   // 768 / 32

__device__ __forceinline__ v8f wmma_f16(const FragAB& a, const FragAB& b, v8f c) {
    // D = A(16x32 f16) * B(32x16 f16) + C(16x16 f32)
    return __builtin_amdgcn_wmma_f32_16x16x32_f16(
        false, a.v, false, b.v, (short)0, c, false, false);
}

// Async global->LDS copy of 16B per lane (CDNA5 GLOBAL_LOAD_ASYNC_TO_LDS_B128,
// tracked by ASYNCcnt). Low 32 bits of a generic pointer into __shared__ are
// the LDS byte address (ISA 10.2: LDS aperture truncates to addr[31:0]).
__device__ __forceinline__ void async_ld_b128(void* lds_ptr, const _Float16* gptr) {
    unsigned lds = (unsigned)(unsigned long long)lds_ptr;
    asm volatile("global_load_async_to_lds_b128 %0, %1, off"
                 :: "v"(lds), "v"(gptr) : "memory");
}
__device__ __forceinline__ void wait_async0() {
    asm volatile("s_wait_asynccnt 0" ::: "memory");
}

// ---------------------------------------------------------------------------
// Kernel 0: f32 -> f16 conversion (hidden and W), vectorized x4.
// ---------------------------------------------------------------------------
__global__ __launch_bounds__(256)
void cvt_f32_f16_kernel(const float* __restrict__ src, _Float16* __restrict__ dst, int n4)
{
    int i = blockIdx.x * blockDim.x + threadIdx.x;
    if (i >= n4) return;
    float4 a = ((const float4*)src)[i];
    v4h o = { (_Float16)a.x, (_Float16)a.y, (_Float16)a.z, (_Float16)a.w };
    ((v4h*)dst)[i] = o;
}

// ---------------------------------------------------------------------------
// Kernel 1: qkv(f16) = hidden_f16(2048x768) @ W_f16^T(768x2304) + bvec
// WG = 256 threads = 8 waves. Tile 128x128, K-step 32, double-buffered LDS
// filled by async global->LDS copies. Wave w: rows (w&3)*32, cols (w>>2)*64.
// ---------------------------------------------------------------------------
__global__ __launch_bounds__(256)
void qkv_gemm_kernel(const _Float16* __restrict__ Ah,
                     const _Float16* __restrict__ Wh,
                     const float* __restrict__ bvec,
                     _Float16* __restrict__ qkvh)
{
    __shared__ __align__(16) _Float16 At[2][128][32]; // A[m][k]
    __shared__ __align__(16) _Float16 Bt[2][128][32]; // Bt[n][k] = W[n][k]

    const int t    = threadIdx.x;
    const int lane = t & 31;
    const int w    = t >> 5;
    const int ln   = lane & 15;
    const int hi   = lane >> 4;

    const int Mbase = blockIdx.y * 128;
    const int Nbase = blockIdx.x * 128;

    const int wr = (w & 3) * 32;
    const int wc = (w >> 2) * 64;

    v8f acc[2][4];
#pragma unroll
    for (int i = 0; i < 2; ++i)
#pragma unroll
        for (int j = 0; j < 4; ++j) acc[i][j] = (v8f){};

    // Stage one 128x32 f16 tile pair into buffer `buf` (pure async copies).
    auto stage = [&](int buf, int k0) {
#pragma unroll
        for (int rep = 0; rep < 2; ++rep) {
            int lin = rep * 256 + t;     // 512 16B-chunks per tile
            int row = lin >> 2;          // 0..127
            int c8  = (lin & 3) << 3;    // 0,8,16,24 (halfs)
            async_ld_b128(&At[buf][row][c8], Ah + (size_t)(Mbase + row) * DIMQ + k0 + c8);
            async_ld_b128(&Bt[buf][row][c8], Wh + (size_t)(Nbase + row) * DIMQ + k0 + c8);
        }
    };

    stage(0, 0);
    for (int ks = 0; ks < NKSTEP; ++ks) {
        const int cur = ks & 1;
        wait_async0();        // my 4 async copies for `cur` are done
        __syncthreads();      // everyone's are done; nobody still reads 1-cur
        if (ks + 1 < NKSTEP) stage(1 - cur, (ks + 1) * 32);  // overlap with compute

        FragAB af[2];
#pragma unroll
        for (int i = 0; i < 2; ++i) {
            int m = wr + 16 * i + ln;
            af[i].h[0] = *(const v8h*)&At[cur][m][8 * hi];
            af[i].h[1] = *(const v8h*)&At[cur][m][16 + 8 * hi];
        }
        FragAB bf[4];
#pragma unroll
        for (int j = 0; j < 4; ++j) {
            int n = wc + 16 * j + ln;
            bf[j].h[0] = *(const v8h*)&Bt[cur][n][8 * hi];
            bf[j].h[1] = *(const v8h*)&Bt[cur][n][16 + 8 * hi];
        }
#pragma unroll
        for (int i = 0; i < 2; ++i)
#pragma unroll
            for (int j = 0; j < 4; ++j)
                acc[i][j] = wmma_f16(af[i], bf[j], acc[i][j]);
    }

    // Epilogue: C layout — lane col = ln (+16*j), rows m = v + 8*hi per VGPR v.
#pragma unroll
    for (int j = 0; j < 4; ++j) {
        int col = Nbase + wc + 16 * j + ln;
        float bb = bvec[col];
#pragma unroll
        for (int i = 0; i < 2; ++i) {
            int mb = Mbase + wr + 16 * i + 8 * hi;
#pragma unroll
            for (int v = 0; v < 8; ++v)
                qkvh[(size_t)(mb + v) * QKVDIM + col] = (_Float16)(acc[i][j][v] + bb);
        }
    }
}

// ---------------------------------------------------------------------------
// Kernel 2: flash attention per (b,h,qtile). WG = 128 threads = 4 waves.
// K tile staged by async global->LDS copies; V staged transposed (f16).
// scores = (Q @ K^T) * 0.125 + bias; online softmax; out = P@V / rowsum.
// ---------------------------------------------------------------------------
__global__ __launch_bounds__(128)
void flash_attn_kernel(const _Float16* __restrict__ qkvh,
                       const float* __restrict__ bias,
                       float* __restrict__ out)
{
    __shared__ __align__(16) _Float16 Kt[32][64];     // Kt[key][d]
    __shared__ __align__(16) _Float16 Vt[64][32];     // Vt[d][key] (transposed)
    __shared__ __align__(16) _Float16 Pt[4][16][32];  // per-wave P tile

    const int t    = threadIdx.x;
    const int lane = t & 31;
    const int w    = t >> 5;
    const int ln   = lane & 15;
    const int hi   = lane >> 4;

    const int qt = blockIdx.x;   // 0..7
    const int h  = blockIdx.y;   // 0..11
    const int b  = blockIdx.z;   // 0..3

    const int    q0      = qt * 64 + w * 16;
    const size_t rowBase = (size_t)b * LB;

    // Q fragments: direct f16 global loads (two K=32 slices of d).
    FragAB qa[2];
    {
        const _Float16* qrow = qkvh + (rowBase + q0 + ln) * QKVDIM + h * HD;
#pragma unroll
        for (int f = 0; f < 2; ++f) {
            qa[f].h[0] = *(const v8h*)(qrow + 32 * f + 8 * hi);
            qa[f].h[1] = *(const v8h*)(qrow + 32 * f + 16 + 8 * hi);
        }
    }

    float mrow[8], lsum[8];
    v8f acc[4];
#pragma unroll
    for (int v = 0; v < 8; ++v) { mrow[v] = -1e30f; lsum[v] = 0.0f; }
#pragma unroll
    for (int j = 0; j < 4; ++j) acc[j] = (v8f){};

    const float* biasB = bias + ((size_t)(b * NH + h)) * SB * SB;

    for (int j0 = 0; j0 < LB; j0 += 32) {
        __syncthreads();   // previous iteration's consumers are done
        // K tile: pure async copies.  V tile: transposed f16 staging.
#pragma unroll
        for (int rep = 0; rep < 2; ++rep) {
            int lin = rep * 128 + t;     // 256 16B-chunks
            int r   = lin >> 3;          // key 0..31
            int c8  = (lin & 7) << 3;    // d 0,8,...,56
            const _Float16* base = qkvh + (rowBase + j0 + r) * QKVDIM + h * HD;
            async_ld_b128(&Kt[r][c8], base + DIMQ + c8);
            v8h vv = *(const v8h*)(base + 2 * DIMQ + c8);
#pragma unroll
            for (int q = 0; q < 8; ++q) Vt[c8 + q][r] = vv[q];
        }
        wait_async0();
        __syncthreads();

        // Scores for 32 keys: two 16x16 C-frags, each = 2 chained WMMAs over d.
        v8f s[2];
#pragma unroll
        for (int tt = 0; tt < 2; ++tt) {
            int kl = 16 * tt + ln;
            FragAB kb0, kb1;
            kb0.h[0] = *(const v8h*)&Kt[kl][8 * hi];
            kb0.h[1] = *(const v8h*)&Kt[kl][16 + 8 * hi];
            kb1.h[0] = *(const v8h*)&Kt[kl][32 + 8 * hi];
            kb1.h[1] = *(const v8h*)&Kt[kl][48 + 8 * hi];
            v8f z = (v8f){};
            z = wmma_f16(qa[0], kb0, z);
            z = wmma_f16(qa[1], kb1, z);
            s[tt] = z;
        }
        // Scale (1/sqrt(64), exact) + bias.
#pragma unroll
        for (int tt = 0; tt < 2; ++tt) {
            int sk = j0 + 16 * tt + ln;
            s[tt] *= 0.125f;
#pragma unroll
            for (int v = 0; v < 8; ++v) {
                int sq = q0 + v + 8 * hi;
                s[tt][v] += biasB[(size_t)sq * SB + sk];
            }
        }
        // Online softmax (row m = v + 8*hi lives across the lane's 16-group).
#pragma unroll
        for (int v = 0; v < 8; ++v) {
            float tm = fmaxf(s[0][v], s[1][v]);
#pragma unroll
            for (int off = 1; off < 16; off <<= 1)
                tm = fmaxf(tm, __shfl_xor(tm, off, 32));
            float mn = fmaxf(mrow[v], tm);
            float cs = __expf(mrow[v] - mn);
            float p0 = __expf(s[0][v] - mn);
            float p1 = __expf(s[1][v] - mn);
            float rs = p0 + p1;
#pragma unroll
            for (int off = 1; off < 16; off <<= 1)
                rs += __shfl_xor(rs, off, 32);
            lsum[v] = lsum[v] * cs + rs;
            mrow[v] = mn;
            s[0][v] = p0; s[1][v] = p1;
#pragma unroll
            for (int j = 0; j < 4; ++j) acc[j][v] *= cs;
        }
        // Transpose P (C layout -> A layout) through per-wave LDS tile.
#pragma unroll
        for (int tt = 0; tt < 2; ++tt)
#pragma unroll
            for (int v = 0; v < 8; ++v)
                Pt[w][v + 8 * hi][ln + 16 * tt] = (_Float16)s[tt][v];

        FragAB pa;
        pa.h[0] = *(const v8h*)&Pt[w][ln][8 * hi];
        pa.h[1] = *(const v8h*)&Pt[w][ln][16 + 8 * hi];

        // P(16x32) @ V(32x64): 4 d-tiles; B-frags contiguous from Vt.
#pragma unroll
        for (int j = 0; j < 4; ++j) {
            FragAB vb;
            vb.h[0] = *(const v8h*)&Vt[16 * j + ln][8 * hi];
            vb.h[1] = *(const v8h*)&Vt[16 * j + ln][16 + 8 * hi];
            acc[j] = wmma_f16(pa, vb, acc[j]);
        }
    }

    // Normalize and store: out[(b*L + q)][h*64 + d], f32.
#pragma unroll
    for (int j = 0; j < 4; ++j) {
#pragma unroll
        for (int v = 0; v < 8; ++v) {
            int ql = q0 + v + 8 * hi;
            float o = acc[j][v] / lsum[v];
            out[(size_t)(b * LB + ql) * DIMQ + h * HD + 16 * j + ln] = o;
        }
    }
}

extern "C" void kernel_launch(void* const* d_in, const int* in_sizes, int n_in,
                              void* d_out, int out_size, void* d_ws, size_t ws_size,
                              hipStream_t stream) {
    const float* hidden = (const float*)d_in[0];   // (2048, 768)
    const float* W      = (const float*)d_in[1];   // (2304, 768)
    const float* bvec   = (const float*)d_in[2];   // (2304,)
    const float* bias   = (const float*)d_in[3];   // (4, 12, 1024, 1024)
    (void)in_sizes; (void)n_in; (void)out_size; (void)ws_size;

    // Workspace layout (f16): hidden | W | qkv  (~16.1 MB total)
    _Float16* Ah   = (_Float16*)d_ws;
    _Float16* Wh   = Ah + (size_t)2048 * DIMQ;
    _Float16* qkvh = Wh + (size_t)QKVDIM * DIMQ;
    float*    outp = (float*)d_out;

    cvt_f32_f16_kernel<<<(2048 * DIMQ / 4 + 255) / 256, 256, 0, stream>>>(
        hidden, Ah, 2048 * DIMQ / 4);
    cvt_f32_f16_kernel<<<(QKVDIM * DIMQ / 4 + 255) / 256, 256, 0, stream>>>(
        W, Wh, QKVDIM * DIMQ / 4);
    qkv_gemm_kernel<<<dim3(QKVDIM / 128, 2048 / 128), 256, 0, stream>>>(
        Ah, Wh, bvec, qkvh);
    flash_attn_kernel<<<dim3(LB / 64, NH, NB), 128, 0, stream>>>(
        qkvh, bias, outp);
}
